// IntegratorNeuronLayer_70085276336407
// MI455X (gfx1250) — matrix operand order repaired and checked by reading, not compiled, not run on hardware
//
#include <hip/hip_runtime.h>

typedef __attribute__((ext_vector_type(16))) __bf16 bf16x16;
typedef __attribute__((ext_vector_type(8)))  float  f32x8;

#define BROWS   131072
#define D_H     1024
#define DIN     1032       // 1024 + 4 + 4
#define H_DIM   64
#define NCOLS   256        // 4 MLPs * 64 hidden
#define NTILES  16         // 256 / 16
#define KSTEPS  33         // ceil(1032/32) -> K padded to 1056
#define HPITCH  264        // 256 + 8 bf16 pad (LDS bank spread)

__device__ __forceinline__ unsigned short f2bf(float f) {
    unsigned u = __builtin_bit_cast(unsigned, f);
    unsigned r = u + 0x7fffu + ((u >> 16) & 1u);   // RNE
    return (unsigned short)(r >> 16);
}
__device__ __forceinline__ float bf2f(unsigned short s) {
    return __builtin_bit_cast(float, (unsigned)s << 16);
}
__device__ __forceinline__ float sigmoidf(float x) {
    return 1.0f / (1.0f + __expf(-x));
}
__device__ __forceinline__ float softplusf(float x) {
    return (x > 20.0f) ? x : log1pf(__expf(x));
}

// ---------------------------------------------------------------------------
// Pack W1cat [1056(Kpad) x 256] (f32, 4 separate (DIN,H) row-major mats) into
// bf16 WMMA-B fragment order: [kstep][ntile][lane][16 bf16].
// lane<16: N=lane,  K = kstep*32 + 0..15 ; lane>=16: N=lane-16, K = +16..31
// ---------------------------------------------------------------------------
__global__ void pack_w1(const float* __restrict__ w1a, const float* __restrict__ w1b,
                        const float* __restrict__ w1g, const float* __restrict__ w1v,
                        unsigned short* __restrict__ wp) {
    int idx = blockIdx.x * blockDim.x + threadIdx.x;
    const int total = KSTEPS * NTILES * 32 * 16;
    if (idx >= total) return;
    int i     = idx & 15;
    int lane  = (idx >> 4) & 31;
    int ntile = (idx >> 9) & 15;
    int kstep = idx >> 13;
    int K   = kstep * 32 + ((lane & 16) ? 16 : 0) + i;
    int col = ntile * 16 + (lane & 15);
    const float* src = (col < 64) ? w1a : (col < 128) ? w1b : (col < 192) ? w1g : w1v;
    int hcol = col & 63;
    float val = (K < DIN) ? src[K * H_DIM + hcol] : 0.0f;
    wp[idx] = f2bf(val);
}

// ---------------------------------------------------------------------------
// Fused: layer1 GEMM (bf16 WMMA) + ReLU -> LDS -> layer2 + integrator
// Block: 256 threads (8 waves), 64 rows. Wave w: row-group w>>1, N-half w&1.
// ---------------------------------------------------------------------------
__global__ __launch_bounds__(256) void integrator_fused(
    const float* __restrict__ h, const float* __restrict__ x, const float* __restrict__ v,
    const unsigned short* __restrict__ wp,
    const float* __restrict__ b1a, const float* __restrict__ b1b,
    const float* __restrict__ b1g, const float* __restrict__ b1v,
    const float* __restrict__ w2a, const float* __restrict__ b2a,
    const float* __restrict__ w2b, const float* __restrict__ b2b,
    const float* __restrict__ w2g, const float* __restrict__ b2g,
    const float* __restrict__ w2v, const float* __restrict__ b2v,
    const float* __restrict__ mu, const float* __restrict__ ex_gamma,
    const float* __restrict__ ex_phi,
    float* __restrict__ x_out, float* __restrict__ v_out)
{
    __shared__ __align__(16) unsigned short hid[64 * HPITCH];  // 33,792 B

    const int tid    = threadIdx.x;
    const int lane   = tid & 31;
    const int wave   = tid >> 5;
    const int rowgrp = wave >> 1;      // 0..3 -> 16-row group
    const int nhalf  = wave & 1;       // 0..1 -> which 8 N-tiles
    const int base   = blockIdx.x * 64;
    const int arow   = base + rowgrp * 16 + (lane & 15);   // A-matrix row this lane holds
    const int koff   = (lane & 16) ? 16 : 0;               // A layout K offset per half-wave

    f32x8 acc[8];
#pragma unroll
    for (int j = 0; j < 8; ++j) acc[j] = (f32x8){0,0,0,0,0,0,0,0};

    const float* hrow = h + (size_t)arow * D_H;
    const bf16x16* bbase = (const bf16x16*)(wp) + (nhalf * 8) * 32 + lane;

    // ---- main K loop over h (1024 cols = 32 steps of K=32) ----
    for (int kstep = 0; kstep < 32; ++kstep) {
        // A: four float4 loads = this lane's share of the rows' 128B line (HBM stream)
        const float* p = hrow + kstep * 32 + koff;
        float4 f0 = *(const float4*)(p);
        float4 f1 = *(const float4*)(p + 4);
        float4 f2 = *(const float4*)(p + 8);
        float4 f3 = *(const float4*)(p + 12);

        // B: all 8 fragments up front (one clause, L0/L2 resident)
        const bf16x16* bptr = bbase + kstep * (NTILES * 32);
        bf16x16 bfrag[8];
#pragma unroll
        for (int j = 0; j < 8; ++j) bfrag[j] = bptr[j * 32];

        float fa[16] = {f0.x, f0.y, f0.z, f0.w, f1.x, f1.y, f1.z, f1.w,
                        f2.x, f2.y, f2.z, f2.w, f3.x, f3.y, f3.z, f3.w};
        bf16x16 afrag;
#pragma unroll
        for (int i = 0; i < 16; ++i) afrag[i] = (__bf16)fa[i];

#pragma unroll
        for (int j = 0; j < 8; ++j) {
            acc[j] = __builtin_amdgcn_wmma_f32_16x16x32_bf16(
                false, afrag, false, bfrag[j], (short)0, acc[j], false, false);
        }
    }

    // ---- K tail: cols 1024..1055 = [x(4), v(4), zeros(24)] ----
    {
        const bf16x16* bptr = bbase + 32 * (NTILES * 32);
        bf16x16 bfrag[8];
#pragma unroll
        for (int j = 0; j < 8; ++j) bfrag[j] = bptr[j * 32];

        float fa[16];
#pragma unroll
        for (int i = 0; i < 16; ++i) fa[i] = 0.0f;
        if (lane < 16) {   // lanes 0-15 hold K 1024..1031 (x,v) ; 1032..1039 are pad 0
            const float4 xr = *(const float4*)(x + (size_t)arow * 4);
            const float4 vr = *(const float4*)(v + (size_t)arow * 4);
            fa[0] = xr.x; fa[1] = xr.y; fa[2] = xr.z; fa[3] = xr.w;
            fa[4] = vr.x; fa[5] = vr.y; fa[6] = vr.z; fa[7] = vr.w;
        }
        bf16x16 afrag;
#pragma unroll
        for (int i = 0; i < 16; ++i) afrag[i] = (__bf16)fa[i];

#pragma unroll
        for (int j = 0; j < 8; ++j) {
            acc[j] = __builtin_amdgcn_wmma_f32_16x16x32_bf16(
                false, afrag, false, bfrag[j], (short)0, acc[j], false, false);
        }
    }

    // ---- bias + ReLU, stash hidden (bf16) into LDS ----
    // C layout: VGPR r, lanes 0-15: M=r, N=lane ; lanes 16-31: M=r+8, N=lane-16
#pragma unroll
    for (int j = 0; j < 8; ++j) {
        const int t   = nhalf * 8 + j;
        const int col = t * 16 + (lane & 15);
        const float* b1p = (col < 64) ? b1a : (col < 128) ? b1b : (col < 192) ? b1g : b1v;
        const float bias = b1p[col & 63];
        const int rbase = rowgrp * 16 + ((lane & 16) ? 8 : 0);
#pragma unroll
        for (int r = 0; r < 8; ++r) {
            float val = acc[j][r] + bias;
            val = val > 0.0f ? val : 0.0f;
            hid[(rbase + r) * HPITCH + col] = f2bf(val);
        }
    }
    __syncthreads();

    // ---- layer 2 + integrator: thread = (row in block, o) ----
    {
        const int rl  = tid >> 2;      // 0..63
        const int o   = tid & 3;
        const int row = base + rl;

        const float* w2s[4] = {w2a, w2b, w2g, w2v};
        float s[4] = {b2a[o], b2b[o], b2g[o], b2v[o]};
#pragma unroll
        for (int role = 0; role < 4; ++role) {
            const float* w2 = w2s[role];
            const uint4* hq = (const uint4*)(hid + rl * HPITCH + role * 64);
            float d = 0.0f;
#pragma unroll
            for (int q = 0; q < 8; ++q) {     // 8 * uint4 = 32 dwords = 64 bf16
                uint4 u = hq[q];
                const unsigned uu[4] = {u.x, u.y, u.z, u.w};
#pragma unroll
                for (int k = 0; k < 4; ++k) {
                    int c = q * 8 + k * 2;
                    d += bf2f((unsigned short)(uu[k] & 0xffffu)) * w2[c * 4 + o];
                    d += bf2f((unsigned short)(uu[k] >> 16))     * w2[(c + 1) * 4 + o];
                }
            }
            s[role] += d;
        }

        const float4 xr = *(const float4*)(x + (size_t)row * 4);
        const float xs[4] = {xr.x, xr.y, xr.z, xr.w};
        float e0 = xs[0] - mu[0], e1 = xs[1] - mu[1],
              e2 = xs[2] - mu[2], e3 = xs[3] - mu[3];
        float imbalance = sqrtf(e0 * e0 + e1 * e1 + e2 * e2 + e3 * e3);

        float vo    = v[(size_t)row * 4 + o];
        float alpha = sigmoidf(s[0]) * __expf(-imbalance);   // KAPPA = 1
        float beta  = softplusf(s[1]);
        float gate  = sigmoidf(s[2]);
        float vc    = s[3];
        float err   = xs[o] - mu[o];
        float harm  = 0.03f * __sinf(ex_gamma[o] * 0.0f + ex_phi[o]);  // t = 0

        float vn = alpha * vo + (1.0f - alpha) * vc - beta * err + harm;
        float xn = xs[o] + 0.1f * gate * vn;                 // DT=0.1, VEL_SCALE=1

        x_out[(size_t)row * 4 + o] = xn;
        v_out[(size_t)row * 4 + o] = vn;
    }
}

extern "C" void kernel_launch(void* const* d_in, const int* in_sizes, int n_in,
                              void* d_out, int out_size, void* d_ws, size_t ws_size,
                              hipStream_t stream) {
    const float* h   = (const float*)d_in[0];
    const float* x   = (const float*)d_in[1];
    const float* v   = (const float*)d_in[2];
    const float* w1a = (const float*)d_in[3];
    const float* b1a = (const float*)d_in[4];
    const float* w2a = (const float*)d_in[5];
    const float* b2a = (const float*)d_in[6];
    const float* w1b = (const float*)d_in[7];
    const float* b1b = (const float*)d_in[8];
    const float* w2b = (const float*)d_in[9];
    const float* b2b = (const float*)d_in[10];
    const float* w1g = (const float*)d_in[11];
    const float* b1g = (const float*)d_in[12];
    const float* w2g = (const float*)d_in[13];
    const float* b2g = (const float*)d_in[14];
    const float* w1v = (const float*)d_in[15];
    const float* b1v = (const float*)d_in[16];
    const float* w2v = (const float*)d_in[17];
    const float* b2v = (const float*)d_in[18];
    const float* mu  = (const float*)d_in[19];
    const float* exg = (const float*)d_in[20];
    const float* exp_ = (const float*)d_in[21];

    unsigned short* wp = (unsigned short*)d_ws;          // 540,672 B packed W1
    float* x_out = (float*)d_out;
    float* v_out = x_out + (size_t)BROWS * 4;

    const int packN = KSTEPS * NTILES * 32 * 16;         // 270,336
    pack_w1<<<(packN + 255) / 256, 256, 0, stream>>>(w1a, w1b, w1g, w1v, wp);

    integrator_fused<<<BROWS / 64, 256, 0, stream>>>(
        h, x, v, wp,
        b1a, b1b, b1g, b1v,
        w2a, b2a, w2b, b2b, w2g, b2g, w2v, b2v,
        mu, exg, exp_, x_out, v_out);
}
